// IMULSTM_14637248545437
// MI455X (gfx1250) — compile-verified, matrix operand order checked by reading, not run on hardware
//
#include <hip/hip_runtime.h>
#include <hip/hip_bf16.h>

// ---------------------------------------------------------------------------
// 2-layer LSTM (B=256, T=256, H=512) + FC/ReLU for MI455X (gfx1250, wave32).
//
// Round 2: fuse both layers into ONE persistent kernel with a software
// pipeline: at global step s, layer-0 WGs compute t=s while layer-1 WGs
// compute t=s-1 (consuming out0[s-1] produced the step before). The
// sequential chain drops from 512 grid barriers to 257, and both per-step
// GEMMs run concurrently (512 WMMA waves/step).
//
//  * 64 persistent WGs x 256 thr (8 waves): WG 0-31 = layer 0, 32-63 = layer 1
//  * bf16 v_wmma_f32_16x16x32_bf16, f32 accumulation
//  * packed [W_hh | W_ih] bf16 weights -> one K-loop/step, bias in acc init
//  * cell state c in LDS; h double-buffered through L2 (weights L2-resident)
// ---------------------------------------------------------------------------

typedef __attribute__((ext_vector_type(16))) __bf16 v16bf;
typedef __attribute__((ext_vector_type(8)))  float  v8f;

#define BB   256     // batch
#define TT   256     // time steps
#define HH   512     // hidden
#define NG   (4*HH)  // 2048 gate rows
#define KP0  544     // packed K, layer 0: 512 (h) + 32 (x padded from 12)
#define KP1  1024    // packed K, layer 1: 512 (h) + 512 (layer-0 out)
#define NWGT 64      // persistent workgroups total (32 per layer)
#define NTHR 256     // 8 waves

// ---- workspace layout (bytes) ---------------------------------------------
#define O_SYNC   ((size_t)0)                         // 256 B (count @0, gen @128)
#define O_H0     ((size_t)256)                       // h buf layer0 [2][B][H] bf16
#define O_XBF    (O_H0   + (size_t)2*BB*HH*2)        // x tile [2][B][32] bf16
#define O_H1     (O_XBF  + (size_t)2*BB*32*2)        // h buf layer1 [2][B][H] bf16
#define O_ZEND   (O_H1   + (size_t)2*BB*HH*2)        // end of zeroed region
#define O_BIAS0  (O_ZEND)                            // [2048] f32
#define O_BIAS1  (O_BIAS0 + (size_t)NG*4)            // [2048] f32
#define O_W0     (O_BIAS1 + (size_t)NG*4)            // [2048][544] bf16
#define O_W1     (O_W0    + (size_t)NG*KP0*2)        // [2048][1024] bf16
#define O_FCW    (O_W1    + (size_t)NG*KP1*2)        // [512][512] bf16
#define O_OUT0   (O_FCW   + (size_t)HH*HH*2)         // [T][B][H] bf16 (64 MB)
#define WS_NEED  (O_OUT0  + (size_t)TT*BB*HH*2)      // ~75 MB total

__device__ __forceinline__ float sigm(float x) {
  return 1.0f / (1.0f + __expf(-x));
}

// device-wide barrier: generation counter, s_sleep backoff
__device__ __forceinline__ void grid_barrier(int* cnt, int* gen, int nblk) {
  __syncthreads();
  if (threadIdx.x == 0) {
    __threadfence();
    int g = __hip_atomic_load(gen, __ATOMIC_ACQUIRE, __HIP_MEMORY_SCOPE_AGENT);
    int prev = __hip_atomic_fetch_add(cnt, 1, __ATOMIC_ACQ_REL, __HIP_MEMORY_SCOPE_AGENT);
    if (prev == nblk - 1) {
      __hip_atomic_store(cnt, 0, __ATOMIC_RELAXED, __HIP_MEMORY_SCOPE_AGENT);
      __hip_atomic_fetch_add(gen, 1, __ATOMIC_RELEASE, __HIP_MEMORY_SCOPE_AGENT);
    } else {
      while (__hip_atomic_load(gen, __ATOMIC_ACQUIRE, __HIP_MEMORY_SCOPE_AGENT) == g)
        __builtin_amdgcn_s_sleep(1);
    }
  }
  __syncthreads();
}

// ---- weight prepack: fp32 -> bf16, [W_hh | W_ih] concat, bias fuse --------
__global__ void __launch_bounds__(256)
prepack_kernel(const float* __restrict__ W_ih0, const float* __restrict__ W_hh0,
               const float* __restrict__ b_ih0, const float* __restrict__ b_hh0,
               const float* __restrict__ W_ih1, const float* __restrict__ W_hh1,
               const float* __restrict__ b_ih1, const float* __restrict__ b_hh1,
               const float* __restrict__ fc_w,
               __bf16* __restrict__ wp0, __bf16* __restrict__ wp1,
               float* __restrict__ bias0, float* __restrict__ bias1,
               __bf16* __restrict__ fcw)
{
  const int N0 = NG * KP0, N1 = NG * KP1, N2 = HH * HH;
  const int total = N0 + N1 + N2 + NG + NG;
  for (int i = blockIdx.x * blockDim.x + threadIdx.x; i < total;
       i += gridDim.x * blockDim.x) {
    if (i < N0) {
      int row = i / KP0, k = i % KP0;
      float v = (k < HH) ? W_hh0[row * HH + k]
                         : ((k < HH + 12) ? W_ih0[row * 12 + (k - HH)] : 0.0f);
      wp0[i] = (__bf16)v;
    } else if (i < N0 + N1) {
      int j = i - N0, row = j / KP1, k = j % KP1;
      float v = (k < HH) ? W_hh1[row * HH + k] : W_ih1[row * HH + (k - HH)];
      wp1[j] = (__bf16)v;
    } else if (i < N0 + N1 + N2) {
      int j = i - N0 - N1;
      fcw[j] = (__bf16)fc_w[j];
    } else if (i < N0 + N1 + N2 + NG) {
      int j = i - N0 - N1 - N2;
      bias0[j] = b_ih0[j] + b_hh0[j];
    } else {
      int j = i - N0 - N1 - N2 - NG;
      bias1[j] = b_ih1[j] + b_hh1[j];
    }
  }
}

// ---- one LSTM timestep for one WG (16 hidden cols, all 256 batch rows) ----
// gates[256 x 64] = [h | u] @ [W_hh | W_ih]^T + bias  via 8 WMMA accumulators
// (2 M-tiles x 4 gates) per wave, K = KP0/KP1.
template <int LAYER>
__device__ __forceinline__ void lstm_step(
    int t, int hc0, int wid, int l16, int lhalf,
    const __bf16* __restrict__ wpack,      // [2048][KP]
    const float bsc[4],                    // per-lane gate biases
    const __bf16* __restrict__ hA,         // h_{t-1}  [B][H]
    const __bf16* __restrict__ uA,         // u_t      [B][ustride]
    int ustride,
    __bf16* __restrict__ hN,               // h_t      [B][H]
    __bf16* __restrict__ out0w,            // layer 0: out0[t] base, else unused
    float* __restrict__ c_lds)             // [B][16] cell state
{
  constexpr int KP = (LAYER == 0) ? KP0 : KP1;
  const int Mbase0 = wid * 32;

  v8f acc[2][4];
#pragma unroll
  for (int mt = 0; mt < 2; ++mt)
#pragma unroll
    for (int g = 0; g < 4; ++g)
#pragma unroll
      for (int r = 0; r < 8; ++r) acc[mt][g][r] = bsc[g];

#pragma unroll 4
  for (int kk = 0; kk < KP; kk += 32) {
    const int k = kk + 16 * lhalf;
    v16bf a[2];
#pragma unroll
    for (int mt = 0; mt < 2; ++mt) {
      const int brow = Mbase0 + mt * 16 + l16;
      const __bf16* ap = (kk < HH)
          ? (hA + (size_t)brow * HH + k)
          : (uA + (size_t)brow * ustride + (k - HH));
      a[mt] = *(const v16bf*)ap;
    }
    v16bf bm[4];
#pragma unroll
    for (int g = 0; g < 4; ++g) {
      const int row = g * HH + hc0 + l16;
      bm[g] = *(const v16bf*)(wpack + (size_t)row * KP + k);
    }
#pragma unroll
    for (int mt = 0; mt < 2; ++mt)
#pragma unroll
      for (int g = 0; g < 4; ++g)
        acc[mt][g] = __builtin_amdgcn_wmma_f32_16x16x32_bf16(
            false, a[mt], false, bm[g], (short)0, acc[mt][g], false, false);
  }

  // gate activations + state update
#pragma unroll
  for (int mt = 0; mt < 2; ++mt) {
#pragma unroll
    for (int r = 0; r < 8; ++r) {
      const int m = Mbase0 + mt * 16 + r + 8 * lhalf;  // batch row
      float iv = sigm(acc[mt][0][r]);
      float fv = sigm(acc[mt][1][r]);
      float gv = tanhf(acc[mt][2][r]);
      float ov = sigm(acc[mt][3][r]);
      float c_old = c_lds[m * 16 + l16];
      float c_new = fv * c_old + iv * gv;
      c_lds[m * 16 + l16] = c_new;
      float hv = ov * tanhf(c_new);
      __bf16 hb = (__bf16)hv;
      hN[(size_t)m * HH + hc0 + l16] = hb;
      if (LAYER == 0)
        out0w[(size_t)m * HH + hc0 + l16] = hb;
    }
  }
}

// ---- fused 2-layer persistent pipeline ------------------------------------
// WG 0..31  -> layer 0, hidden cols [wg*16, wg*16+16)
// WG 32..63 -> layer 1, hidden cols [(wg-32)*16, ...)
// Global step s: layer 0 runs t=s (s<T), layer 1 runs t=s-1 (s>=1).
__global__ void __launch_bounds__(NTHR)
lstm_fused_kernel(const float* __restrict__ x,      // [B][T][12]
                  const __bf16* __restrict__ wp0,   // [2048][KP0]
                  const __bf16* __restrict__ wp1,   // [2048][KP1]
                  const float* __restrict__ bias0,  // [2048]
                  const float* __restrict__ bias1,  // [2048]
                  __bf16* __restrict__ h0b,         // [2][B][H]
                  __bf16* __restrict__ h1b,         // [2][B][H]
                  __bf16* __restrict__ xbf,         // [2][B][32]
                  __bf16* __restrict__ out0,        // [T][B][H]
                  int* cnt, int* gen)
{
  const bool isL0 = blockIdx.x < 32;
  const int  wg   = blockIdx.x & 31;
  const int tid  = threadIdx.x;
  const int wid  = tid >> 5;
  const int lane = tid & 31;
  const int l16  = lane & 15;
  const int lhalf = lane >> 4;
  const int hc0  = wg * 16;

  __shared__ float c_lds[BB * 16];                 // cell state, 16 KB
  for (int i = tid; i < BB * 16; i += NTHR) c_lds[i] = 0.0f;

  const float*  bias  = isL0 ? bias0 : bias1;
  const __bf16* wpack = isL0 ? wp0 : wp1;
  __bf16*       hbuf  = isL0 ? h0b : h1b;

  // per-lane gate biases (constant over t and M; N column = lane&15)
  float bsc[4];
#pragma unroll
  for (int g = 0; g < 4; ++g) bsc[g] = bias[g * HH + hc0 + l16];

  // stage x tile for t=0 (padding cols 12..31 pre-zeroed by host memset)
  if (isL0) {
    int gid = wg * NTHR + tid;
    if (gid < BB * 12) {
      int b = gid / 12, f = gid % 12;
      xbf[b * 32 + f] = (__bf16)x[((size_t)b * TT + 0) * 12 + f];
    }
  }
  __syncthreads();
  grid_barrier(cnt, gen, NWGT);

  for (int s = 0; s <= TT; ++s) {
    if (isL0) {
      if (s < TT) {
        const int t = s, cur = t & 1, nxt = cur ^ 1;
        lstm_step<0>(t, hc0, wid, l16, lhalf, wpack, bsc,
                     h0b + (size_t)cur * BB * HH,
                     xbf + (size_t)cur * BB * 32, 32,
                     h0b + (size_t)nxt * BB * HH,
                     out0 + (size_t)t * BB * HH, c_lds);
        // stage next x tile
        if (t + 1 < TT) {
          int gid = wg * NTHR + tid;
          if (gid < BB * 12) {
            int b = gid / 12, f = gid % 12;
            xbf[(size_t)nxt * BB * 32 + b * 32 + f] =
                (__bf16)x[((size_t)b * TT + (t + 1)) * 12 + f];
          }
        }
      }
    } else {
      if (s >= 1) {
        const int t = s - 1, cur = t & 1, nxt = cur ^ 1;
        lstm_step<1>(t, hc0, wid, l16, lhalf, wpack, bsc,
                     h1b + (size_t)cur * BB * HH,
                     out0 + (size_t)t * BB * HH, HH,
                     h1b + (size_t)nxt * BB * HH,
                     nullptr, c_lds);
      }
    }
    grid_barrier(cnt, gen, NWGT);
  }
}

// ---- FC + ReLU: out = relu(h1 @ fc_w^T + fc_b), 256x512x512 ---------------
__global__ void __launch_bounds__(256)
fc_kernel(const __bf16* __restrict__ h1,   // [B][H] final hidden (bf16)
          const __bf16* __restrict__ wfc,  // [H][H] bf16 row-major [out][in]
          const float* __restrict__ fcb,   // [H]
          float* __restrict__ out)         // [B][H] f32
{
  const int tid = threadIdx.x, wid = tid >> 5, lane = tid & 31;
  const int l16 = lane & 15, lhalf = lane >> 4;
  const int tile = blockIdx.x * 8 + wid;   // 512 tiles: 16 M x 32 N
  const int mt = tile & 15, nt = tile >> 4;
  const int Mb = mt * 16, Nb = nt * 16;

  float bv = fcb[Nb + l16];
  v8f acc;
#pragma unroll
  for (int r = 0; r < 8; ++r) acc[r] = bv;

#pragma unroll 4
  for (int kk = 0; kk < HH; kk += 32) {
    v16bf a = *(const v16bf*)(h1 + (size_t)(Mb + l16) * HH + kk + 16 * lhalf);
    v16bf b = *(const v16bf*)(wfc + (size_t)(Nb + l16) * HH + kk + 16 * lhalf);
    acc = __builtin_amdgcn_wmma_f32_16x16x32_bf16(false, a, false, b, (short)0,
                                                  acc, false, false);
  }
#pragma unroll
  for (int r = 0; r < 8; ++r) {
    int m = Mb + r + 8 * lhalf;
    float v = acc[r];
    out[(size_t)m * HH + Nb + l16] = v > 0.0f ? v : 0.0f;
  }
}

// ---------------------------------------------------------------------------
extern "C" void kernel_launch(void* const* d_in, const int* in_sizes, int n_in,
                              void* d_out, int out_size, void* d_ws, size_t ws_size,
                              hipStream_t stream)
{
  (void)in_sizes; (void)n_in; (void)out_size; (void)ws_size; // needs ~75 MB ws

  const float* x     = (const float*)d_in[0];
  const float* W_ih0 = (const float*)d_in[1];
  const float* W_hh0 = (const float*)d_in[2];
  const float* b_ih0 = (const float*)d_in[3];
  const float* b_hh0 = (const float*)d_in[4];
  const float* W_ih1 = (const float*)d_in[5];
  const float* W_hh1 = (const float*)d_in[6];
  const float* b_ih1 = (const float*)d_in[7];
  const float* b_hh1 = (const float*)d_in[8];
  const float* fc_w  = (const float*)d_in[9];
  const float* fc_b  = (const float*)d_in[10];

  char* ws = (char*)d_ws;
  int*    cnt   = (int*)(ws + O_SYNC);
  int*    gen   = (int*)(ws + O_SYNC + 128);
  __bf16* h0b   = (__bf16*)(ws + O_H0);
  __bf16* xbf   = (__bf16*)(ws + O_XBF);
  __bf16* h1b   = (__bf16*)(ws + O_H1);
  float*  bias0 = (float*)(ws + O_BIAS0);
  float*  bias1 = (float*)(ws + O_BIAS1);
  __bf16* wp0   = (__bf16*)(ws + O_W0);
  __bf16* wp1   = (__bf16*)(ws + O_W1);
  __bf16* fcw   = (__bf16*)(ws + O_FCW);
  __bf16* out0  = (__bf16*)(ws + O_OUT0);

  // zero sync counters, h double-buffers, x staging (incl. zero K-padding)
  hipMemsetAsync(ws, 0, O_ZEND, stream);

  prepack_kernel<<<2048, 256, 0, stream>>>(W_ih0, W_hh0, b_ih0, b_hh0,
                                           W_ih1, W_hh1, b_ih1, b_hh1,
                                           fc_w, wp0, wp1, bias0, bias1, fcw);

  lstm_fused_kernel<<<NWGT, NTHR, 0, stream>>>(x, wp0, wp1, bias0, bias1,
                                               h0b, h1b, xbf, out0, cnt, gen);

  // T=256 even -> final layer-1 hidden state lands in buffer 0 of h1b
  fc_kernel<<<64, 256, 0, stream>>>(h1b, fcw, fc_b, (float*)d_out);
}
